// SAModule_16844861734919
// MI455X (gfx1250) — matrix operand order repaired and checked by reading, not compiled, not run on hardware
//
#include <hip/hip_runtime.h>
#include <hip/hip_bf16.h>
#include <math.h>

#define BCLOUDS 4
#define NP      8192
#define MPTS    4096
#define KNBR    64
#define CIN     64
#define BM      (BCLOUDS*MPTS)
#define RADSQ   0.04f

typedef __attribute__((ext_vector_type(16))) __bf16 v16bf;
typedef __attribute__((ext_vector_type(8)))  __bf16 v8bf;
typedef __attribute__((ext_vector_type(8)))  float  v8f;

static __device__ __forceinline__ v16bf cat8(v8bf lo, v8bf hi) {
  return __builtin_shufflevector(lo, hi, 0,1,2,3,4,5,6,7,8,9,10,11,12,13,14,15);
}

// ---------------------------------------------------------------------------
// Kernel 1: farthest point sampling. One workgroup per cloud; 1024 threads,
// 8 points/thread held in registers; per-step argmax = shuffle-reduce per
// wave, then LDS reduce across 32 waves.
// ---------------------------------------------------------------------------
__global__ __launch_bounds__(1024)
void fps_kernel(const float* __restrict__ pos, int* __restrict__ idxg) {
  const int b = blockIdx.x;
  const float* posc = pos + (size_t)b * NP * 3;
  const int t = threadIdx.x;
  const int lane = t & 31, wid = t >> 5;
  __shared__ float swv[32];
  __shared__ int   swi[32];
  __shared__ float sc[3];

  float px[8], py[8], pz[8], mind[8];
  const int pbase = t * 8;
  float cx = posc[0], cy = posc[1], cz = posc[2];
#pragma unroll
  for (int j = 0; j < 8; ++j) {
    int p = pbase + j;
    px[j] = posc[p*3+0]; py[j] = posc[p*3+1]; pz[j] = posc[p*3+2];
    float dx = px[j]-cx, dy = py[j]-cy, dz = pz[j]-cz;
    mind[j] = dx*dx + dy*dy + dz*dz;
  }
  if (t == 0) idxg[b*MPTS] = 0;

  for (int step = 1; step < MPTS; ++step) {
    float bv = -1.0f; int bi = pbase;
#pragma unroll
    for (int j = 0; j < 8; ++j)
      if (mind[j] > bv) { bv = mind[j]; bi = pbase + j; }
#pragma unroll
    for (int off = 16; off >= 1; off >>= 1) {
      float ov = __shfl_down(bv, off, 32);
      int   oi = __shfl_down(bi, off, 32);
      if (ov > bv) { bv = ov; bi = oi; }
    }
    if (lane == 0) { swv[wid] = bv; swi[wid] = bi; }
    __syncthreads();
    if (t < 32) {
      bv = swv[lane]; bi = swi[lane];
#pragma unroll
      for (int off = 16; off >= 1; off >>= 1) {
        float ov = __shfl_down(bv, off, 32);
        int   oi = __shfl_down(bi, off, 32);
        if (ov > bv) { bv = ov; bi = oi; }
      }
      if (lane == 0) {
        idxg[b*MPTS + step] = bi;
        sc[0] = posc[bi*3+0]; sc[1] = posc[bi*3+1]; sc[2] = posc[bi*3+2];
      }
    }
    __syncthreads();
    cx = sc[0]; cy = sc[1]; cz = sc[2];
#pragma unroll
    for (int j = 0; j < 8; ++j) {
      float dx = px[j]-cx, dy = py[j]-cy, dz = pz[j]-cz;
      mind[j] = fminf(mind[j], dx*dx + dy*dy + dz*dz);
    }
  }
}

// ---------------------------------------------------------------------------
// Kernel 2: gather sampled centroid positions; emit pos_out / batch_out.
// ---------------------------------------------------------------------------
__global__ void gatherq_kernel(const float* __restrict__ pos, const int* __restrict__ idxg,
                               float* __restrict__ qpos, float* __restrict__ posout,
                               float* __restrict__ batchout) {
  int i = blockIdx.x * blockDim.x + threadIdx.x;
  if (i >= BM) return;
  int b = i >> 12;
  int id = idxg[i];
  const float* pp = pos + ((size_t)b * NP + id) * 3;
  float p0 = pp[0], p1 = pp[1], p2 = pp[2];
  qpos[i*3+0] = p0; qpos[i*3+1] = p1; qpos[i*3+2] = p2;
  posout[i*3+0] = p0; posout[i*3+1] = p1; posout[i*3+2] = p2;
  batchout[i] = (float)b;
}

// ---------------------------------------------------------------------------
// Kernel 3: radius query. One wave per centroid; self-edge seeded in slot 0,
// then first up-to-K points within R appended with ballot + prefix popcount.
// ---------------------------------------------------------------------------
__global__ __launch_bounds__(256)
void nbr_kernel(const float* __restrict__ pos, const int* __restrict__ idxg,
                const float* __restrict__ qpos, int* __restrict__ nbr) {
  const int w = threadIdx.x >> 5, lane = threadIdx.x & 31;
  const int c = blockIdx.x * 8 + w;
  const int b = c >> 12;
  const float* posc = pos + (size_t)b * NP * 3;
  float qx = qpos[c*3+0], qy = qpos[c*3+1], qz = qpos[c*3+2];
  int self = idxg[c];
  nbr[c*KNBR + lane]      = -1;
  nbr[c*KNBR + 32 + lane] = -1;
  if (lane == 0) nbr[c*KNBR] = self;
  int count = 1;
  for (int tt = 0; tt < NP/32; ++tt) {
    int p = tt*32 + lane;
    float dx = posc[p*3+0]-qx, dy = posc[p*3+1]-qy, dz = posc[p*3+2]-qz;
    float d2 = dx*dx + dy*dy + dz*dz;
    bool pred = (d2 <= RADSQ) && (p != self);
    unsigned m = __builtin_amdgcn_ballot_w32(pred);
    if (m) {
      int pre  = __popc(m & ((1u << lane) - 1u));
      int slot = count + pre;
      if (pred && slot < KNBR) nbr[c*KNBR + slot] = p;
      count += __popc(m);
      if (count >= KNBR) break;
    }
  }
}

// ---------------------------------------------------------------------------
// Kernel 4: fused PointNetConv. 4 waves/WG, one centroid per wave.
//   layer1: feat[64 x 96(bf16,pad)] @ W1T -> relu -> h1 (LDS bf16)
//   layer2: h1[64 x 64] @ W2T -> relu -> masked max over edges
// All matmuls via v_wmma_f32_16x16x32_bf16.
// ---------------------------------------------------------------------------
__global__ __launch_bounds__(128)
void conv_kernel(const float* __restrict__ x, const float* __restrict__ pos,
                 const float* __restrict__ qpos, const int* __restrict__ nbr,
                 const float* __restrict__ W1, const float* __restrict__ b1,
                 const float* __restrict__ W2, const float* __restrict__ b2,
                 float* __restrict__ xout) {
  __shared__ __align__(16) __bf16 sW1T[64*96];    // [n][k], k padded 67->96
  __shared__ __align__(16) __bf16 sW2T[128*64];   // [n][k]
  __shared__ float sb1[64];
  __shared__ float sb2[128];
  __shared__ __align__(16) __bf16 sh1[4][64*64];  // per-wave h1 tile
  __shared__ float svm[4][64];                    // 0 or -inf per edge
  __shared__ int   spidx[4][64];                  // neighbor point index

  const int tid = threadIdx.x;
  // stage transposed bf16 weights + biases once per workgroup
  for (int i = tid; i < 64*96; i += 128) {
    int n = i / 96, k = i - n*96;
    sW1T[i] = (__bf16)((k < CIN+3) ? W1[k*64 + n] : 0.0f);
  }
  for (int i = tid; i < 128*64; i += 128) {
    int n = i >> 6, k = i & 63;
    sW2T[i] = (__bf16)W2[k*128 + n];
  }
  if (tid < 64) sb1[tid] = b1[tid];
  sb2[tid] = b2[tid];

  const int w = tid >> 5, lane = tid & 31;
  const int c = blockIdx.x * 4 + w;
  const int b = c >> 12;
  const float* posc = pos + (size_t)b * NP * 3;
  const float* xc   = x   + (size_t)b * NP * CIN;
  float qx = qpos[c*3+0], qy = qpos[c*3+1], qz = qpos[c*3+2];

  int e0 = nbr[c*KNBR + lane];
  int e1 = nbr[c*KNBR + 32 + lane];
  spidx[w][lane]      = e0;
  spidx[w][32 + lane] = e1;
  svm[w][lane]        = (e0 < 0) ? -INFINITY : 0.0f;
  svm[w][32 + lane]   = (e1 < 0) ? -INFINITY : 0.0f;
  __syncthreads();

  const int h = lane >> 4, r0 = lane & 15;

  // ---------------- layer 1 ----------------
  for (int mt = 0; mt < 4; ++mt) {
    int row = mt*16 + r0;
    int pt = spidx[w][row]; if (pt < 0) pt = 0;
    const float* xr = xc + (size_t)pt * CIN;
    v16bf a[3];
#pragma unroll
    for (int kc = 0; kc < 2; ++kc) {       // channels 0..63, A-layout runs
      int kb = kc*32 + h*8;
      float4 f0 = *(const float4*)(xr + kb);
      float4 f1 = *(const float4*)(xr + kb + 4);
      float4 f2 = *(const float4*)(xr + kb + 16);
      float4 f3 = *(const float4*)(xr + kb + 20);
      v16bf av;
      av[0]=(__bf16)f0.x;  av[1]=(__bf16)f0.y;  av[2]=(__bf16)f0.z;  av[3]=(__bf16)f0.w;
      av[4]=(__bf16)f1.x;  av[5]=(__bf16)f1.y;  av[6]=(__bf16)f1.z;  av[7]=(__bf16)f1.w;
      av[8]=(__bf16)f2.x;  av[9]=(__bf16)f2.y;  av[10]=(__bf16)f2.z; av[11]=(__bf16)f2.w;
      av[12]=(__bf16)f3.x; av[13]=(__bf16)f3.y; av[14]=(__bf16)f3.z; av[15]=(__bf16)f3.w;
      a[kc] = av;
    }
    {                                      // tail chunk: rel pos at K=64..66
      float rx = 0.f, ry = 0.f, rz = 0.f;
      if (h == 0) {
        rx = posc[pt*3+0] - qx; ry = posc[pt*3+1] - qy; rz = posc[pt*3+2] - qz;
      }
      v16bf av;
#pragma unroll
      for (int i = 0; i < 16; ++i) av[i] = (__bf16)0.0f;
      av[0] = (__bf16)rx; av[1] = (__bf16)ry; av[2] = (__bf16)rz;
      a[2] = av;
    }
#pragma unroll
    for (int nt = 0; nt < 4; ++nt) {
      int col = nt*16 + r0;
      float bv = sb1[col];
      v8f acc = {bv,bv,bv,bv,bv,bv,bv,bv};
#pragma unroll
      for (int kc = 0; kc < 3; ++kc) {
        const v8bf* bp = (const v8bf*)&sW1T[col*96 + kc*32 + h*8];
        v16bf bf = cat8(bp[0], bp[2]);
        acc = __builtin_amdgcn_wmma_f32_16x16x32_bf16(false, a[kc], false, bf,
                                                      (short)0, acc, false, false);
      }
#pragma unroll
      for (int v = 0; v < 8; ++v) {
        float rv = fmaxf(acc[v], 0.0f);
        int rr = mt*16 + v + 8*h;
        sh1[w][rr*64 + col] = (__bf16)rv;
      }
    }
  }
  __syncthreads();

  // ---------------- layer 2 + masked max ----------------
  float run[8];
#pragma unroll
  for (int nt = 0; nt < 8; ++nt) run[nt] = -INFINITY;

  for (int mt = 0; mt < 4; ++mt) {
    int row = mt*16 + r0;
    v16bf a2[2];
#pragma unroll
    for (int kc = 0; kc < 2; ++kc) {
      const v8bf* p0 = (const v8bf*)&sh1[w][row*64 + kc*32 + h*8];
      const v8bf* p1 = (const v8bf*)&sh1[w][row*64 + kc*32 + 16 + h*8];
      a2[kc] = cat8(*p0, *p1);
    }
    float vm[8];
#pragma unroll
    for (int v = 0; v < 8; ++v) vm[v] = svm[w][mt*16 + v + 8*h];
#pragma unroll
    for (int nt = 0; nt < 8; ++nt) {
      int col = nt*16 + r0;
      float bv = sb2[col];
      v8f acc = {bv,bv,bv,bv,bv,bv,bv,bv};
#pragma unroll
      for (int kc = 0; kc < 2; ++kc) {
        const v8bf* bp = (const v8bf*)&sW2T[col*64 + kc*32 + h*8];
        v16bf bf = cat8(bp[0], bp[2]);
        acc = __builtin_amdgcn_wmma_f32_16x16x32_bf16(false, a2[kc], false, bf,
                                                      (short)0, acc, false, false);
      }
      float part = -INFINITY;
#pragma unroll
      for (int v = 0; v < 8; ++v)
        part = fmaxf(part, fmaxf(acc[v], 0.0f) + vm[v]);
      run[nt] = fmaxf(run[nt], part);
    }
  }
#pragma unroll
  for (int nt = 0; nt < 8; ++nt) {
    float other = __shfl_xor(run[nt], 16, 32);
    float rmax = fmaxf(run[nt], other);
    if (lane < 16) xout[(size_t)c*128 + nt*16 + lane] = rmax;
  }
}

// ---------------------------------------------------------------------------
extern "C" void kernel_launch(void* const* d_in, const int* in_sizes, int n_in,
                              void* d_out, int out_size, void* d_ws, size_t ws_size,
                              hipStream_t stream) {
  (void)in_sizes; (void)n_in; (void)out_size; (void)ws_size;
  const float* x   = (const float*)d_in[0];
  const float* pos = (const float*)d_in[1];
  // d_in[2] = batch (implicit via layout)
  const float* W1  = (const float*)d_in[3];
  const float* b1  = (const float*)d_in[4];
  const float* W2  = (const float*)d_in[5];
  const float* b2  = (const float*)d_in[6];

  float* xout     = (float*)d_out;                 // [BM,128]
  float* posout   = xout + (size_t)BM * 128;       // [BM,3]
  float* batchout = posout + (size_t)BM * 3;       // [BM]

  int*   idxg = (int*)d_ws;                        // [BM]
  float* qpos = (float*)(idxg + BM);               // [BM,3]
  int*   nbrw = (int*)(qpos + (size_t)BM * 3);     // [BM,KNBR]

  fps_kernel<<<BCLOUDS, 1024, 0, stream>>>(pos, idxg);
  gatherq_kernel<<<BM/256, 256, 0, stream>>>(pos, idxg, qpos, posout, batchout);
  nbr_kernel<<<BM/8, 256, 0, stream>>>(pos, idxg, qpos, nbrw);
  conv_kernel<<<BM/4, 128, 0, stream>>>(x, pos, qpos, nbrw, W1, b1, W2, b2, xout);
}
